// HeteroRGCN_36429912604932
// MI455X (gfx1250) — matrix operand order repaired and checked by reading, not compiled, not run on hardware
//
#include <hip/hip_runtime.h>
#include <hip/hip_bf16.h>

// ---------------------------------------------------------------------------
// HeteroRGCN (2-layer) for gfx1250 / MI455X.
//   - Node transforms x@W+b via v_wmma_f32_16x16x32_f16 (f32 accumulate).
//   - Edge scatter-mean via float4 gather + global_atomic_add_f32.
//   - Layer 2 computes only the target-node path (h_c/h_d of layer 2 are dead).
// ---------------------------------------------------------------------------

typedef __attribute__((ext_vector_type(16))) _Float16 v16h;
typedef __attribute__((ext_vector_type(8)))  float    v8f;

#define LDS_KPAD 72  // halves per transposed-weight row: 64 + 8 pad (bank-conflict free)

// Y[r, 0:64] = X[r, 0:64] @ W[64,64] + bias[64]   (all f32 in memory, f16 WMMA inputs)
// One wave -> 16 rows x 64 cols. Block = 256 threads = 8 waves -> 128 rows/block.
__global__ __launch_bounds__(256)
void gemm64_wmma_kernel(const float* __restrict__ X, const float* __restrict__ W,
                        const float* __restrict__ bias, float* __restrict__ Y, int nrows)
{
    __shared__ _Float16 Wt[64 * LDS_KPAD];  // Wt[n][k] = (f16) W[k][n]
    const int tid = threadIdx.x;

    // Stage weight, transposed, f32 -> f16 (coalesced global reads).
    for (int idx = tid; idx < 64 * 64; idx += 256) {
        const int k = idx >> 6, n = idx & 63;
        Wt[n * LDS_KPAD + k] = (_Float16)W[idx];
    }
    __syncthreads();

    const int wave  = tid >> 5;
    const int lane  = tid & 31;
    const int lhalf = lane >> 4;   // lane group 0/1
    const int lmod  = lane & 15;

    const int row_base = (blockIdx.x * 8 + wave) * 16;
    if (row_base >= nrows) return;  // wave-uniform exit; EXEC stays all-ones below

    // Preload B fragments [kiter][ntile] from LDS.
    // Layout (16-bit B 32x16, contiguous k order used here, not the ISA table order): N = lane%16, K = (lane/16)*16 + i  (i = vec elem).
    v16h Bf[2][4];
#pragma unroll
    for (int ki = 0; ki < 2; ++ki) {
        const int kb = ki * 32 + lhalf * 16;
#pragma unroll
        for (int nt = 0; nt < 4; ++nt) {
            const _Float16* p = &Wt[(nt * 16 + lmod) * LDS_KPAD + kb];
            v16h b;
#pragma unroll
            for (int i = 0; i < 16; ++i) b[i] = p[i];
            Bf[ki][nt] = b;
        }
    }

    // Accumulators initialized with bias (C/D layout: N = lane%16, M = r + (lane/16)*8).
    v8f acc[4];
#pragma unroll
    for (int nt = 0; nt < 4; ++nt) {
        const float bv = bias[nt * 16 + lmod];
        v8f c;
#pragma unroll
        for (int r = 0; r < 8; ++r) c[r] = bv;
        acc[nt] = c;
    }

    // A operand: lane holds row = row_base + lane%16; clamp tail rows (stores masked later).
    int row = row_base + lmod;
    if (row >= nrows) row = nrows - 1;
    const float* __restrict__ xr = X + (size_t)row * 64;

#pragma unroll
    for (int ki = 0; ki < 2; ++ki) {
        // A layout (matches B): elems 0..15 -> K = k0..k0+15,
        // with k0 = ki*32 + (lane/16)*16.
        const int k0 = ki * 32 + lhalf * 16;
        const float4 c0 = *(const float4*)(xr + k0);
        const float4 c1 = *(const float4*)(xr + k0 + 4);
        const float4 c2 = *(const float4*)(xr + k0 + 8);
        const float4 c3 = *(const float4*)(xr + k0 + 12);
        v16h a;
        a[0]  = (_Float16)c0.x; a[1]  = (_Float16)c0.y; a[2]  = (_Float16)c0.z; a[3]  = (_Float16)c0.w;
        a[4]  = (_Float16)c1.x; a[5]  = (_Float16)c1.y; a[6]  = (_Float16)c1.z; a[7]  = (_Float16)c1.w;
        a[8]  = (_Float16)c2.x; a[9]  = (_Float16)c2.y; a[10] = (_Float16)c2.z; a[11] = (_Float16)c2.w;
        a[12] = (_Float16)c3.x; a[13] = (_Float16)c3.y; a[14] = (_Float16)c3.z; a[15] = (_Float16)c3.w;
#pragma unroll
        for (int nt = 0; nt < 4; ++nt)
            acc[nt] = __builtin_amdgcn_wmma_f32_16x16x32_f16(
                false, a, false, Bf[ki][nt], (short)0, acc[nt], false, false);
    }

    // Store D: lane writes (M = r + lhalf*8, N = nt*16 + lmod); mask tail rows.
#pragma unroll
    for (int nt = 0; nt < 4; ++nt) {
#pragma unroll
        for (int r = 0; r < 8; ++r) {
            const int m = row_base + r + lhalf * 8;
            if (m < nrows)
                Y[(size_t)m * 64 + nt * 16 + lmod] = acc[nt][r];
        }
    }
}

// acc[dst] += Wh[src] (16 threads/edge, float4 lanes); deg[dst] += 1
__global__ __launch_bounds__(256)
void scatter_acc_kernel(const float* __restrict__ Wh, const int* __restrict__ src,
                        const int* __restrict__ dst, float* __restrict__ acc,
                        float* __restrict__ deg, int nedges)
{
    const long long t = (long long)blockIdx.x * 256 + threadIdx.x;
    const long long e = t >> 4;
    if (e >= nedges) return;
    const int c = (int)(t & 15);
    const int s = src[e];
    const int d = dst[e];
    const float4 v = *(const float4*)(Wh + (size_t)s * 64 + c * 4);
    float* ap = acc + (size_t)d * 64 + c * 4;
    atomicAdd(ap + 0, v.x);
    atomicAdd(ap + 1, v.y);
    atomicAdd(ap + 2, v.z);
    atomicAdd(ap + 3, v.w);
    if (c == 0) atomicAdd(deg + d, 1.0f);
}

__global__ void zero_kernel(float* __restrict__ p, long long n)
{
    const long long i = (long long)blockIdx.x * 256 + threadIdx.x;
    if (i < n) p[i] = 0.0f;
}

// h = lrelu(acc_a/max(deg_a,1) + acc_b/max(deg_b,1))   (target nodes, layer 1)
__global__ void finalize_sum2_lrelu_kernel(const float* __restrict__ acc_a,
                                           const float* __restrict__ acc_b,
                                           const float* __restrict__ deg_a,
                                           const float* __restrict__ deg_b,
                                           float* __restrict__ h, long long n64)
{
    const long long i = (long long)blockIdx.x * 256 + threadIdx.x;
    if (i >= n64) return;
    const long long node = i >> 6;
    const float ia = 1.0f / fmaxf(deg_a[node], 1.0f);
    const float ib = 1.0f / fmaxf(deg_b[node], 1.0f);
    const float v = acc_a[i] * ia + acc_b[i] * ib;
    h[i] = v > 0.0f ? v : 0.01f * v;
}

// h = lrelu(acc/max(deg,1))   (card/device nodes, layer 1)
__global__ void finalize_lrelu_kernel(const float* __restrict__ acc,
                                      const float* __restrict__ deg,
                                      float* __restrict__ h, long long n64)
{
    const long long i = (long long)blockIdx.x * 256 + threadIdx.x;
    if (i >= n64) return;
    const float v = acc[i] * (1.0f / fmaxf(deg[i >> 6], 1.0f));
    h[i] = v > 0.0f ? v : 0.01f * v;
}

// Layer-2 target mean-sum fused with the 64->2 output projection.
__global__ __launch_bounds__(256)
void finalize_out_kernel(const float* __restrict__ acc_a, const float* __restrict__ acc_b,
                         const float* __restrict__ deg_a, const float* __restrict__ deg_b,
                         const float* __restrict__ Wout, const float* __restrict__ bout,
                         float* __restrict__ out, int n)
{
    __shared__ float w0[64], w1[64];
    const int tid = threadIdx.x;
    if (tid < 64) { w0[tid] = Wout[tid * 2]; w1[tid] = Wout[tid * 2 + 1]; }
    __syncthreads();

    const int node = blockIdx.x * 256 + tid;
    if (node >= n) return;
    const float ia = 1.0f / fmaxf(deg_a[node], 1.0f);
    const float ib = 1.0f / fmaxf(deg_b[node], 1.0f);
    const float* pa = acc_a + (size_t)node * 64;
    const float* pb = acc_b + (size_t)node * 64;
    float o0 = bout[0], o1 = bout[1];
#pragma unroll
    for (int k = 0; k < 64; k += 4) {
        const float4 va = *(const float4*)(pa + k);
        const float4 vb = *(const float4*)(pb + k);
        const float h0 = va.x * ia + vb.x * ib;
        const float h1 = va.y * ia + vb.y * ib;
        const float h2 = va.z * ia + vb.z * ib;
        const float h3 = va.w * ia + vb.w * ib;
        o0 += h0 * w0[k] + h1 * w0[k + 1] + h2 * w0[k + 2] + h3 * w0[k + 3];
        o1 += h0 * w1[k] + h1 * w1[k + 1] + h2 * w1[k + 2] + h3 * w1[k + 3];
    }
    out[(size_t)node * 2 + 0] = o0;
    out[(size_t)node * 2 + 1] = o1;
}

static inline int cdiv_i(long long a, long long b) { return (int)((a + b - 1) / b); }

extern "C" void kernel_launch(void* const* d_in, const int* in_sizes, int n_in,
                              void* d_out, int out_size, void* d_ws, size_t ws_size,
                              hipStream_t stream)
{
    const float* features   = (const float*)d_in[0];
    const float* emb_card   = (const float*)d_in[1];
    const float* emb_device = (const float*)d_in[2];
    const float* W1         = (const float*)d_in[3];
    const float* b1         = (const float*)d_in[4];
    const float* W2         = (const float*)d_in[5];
    const float* b2         = (const float*)d_in[6];
    const float* W_out      = (const float*)d_in[7];
    const float* b_out      = (const float*)d_in[8];
    const int* src_c2t = (const int*)d_in[9];
    const int* dst_c2t = (const int*)d_in[10];
    const int* src_t2c = (const int*)d_in[11];
    const int* dst_t2c = (const int*)d_in[12];
    const int* src_d2t = (const int*)d_in[13];
    const int* dst_d2t = (const int*)d_in[14];
    const int* src_t2d = (const int*)d_in[15];
    const int* dst_t2d = (const int*)d_in[16];
    float* out = (float*)d_out;

    const int N_T = in_sizes[0] / 64;
    const int N_C = in_sizes[1] / 64;
    const int N_D = in_sizes[2] / 64;
    const int E   = in_sizes[9];

    // Workspace layout (floats). Zeroed regions are contiguous prefixes.
    float* ws = (float*)d_ws;
    size_t off = 0;
    float* acc_t_a = ws + off; off += (size_t)N_T * 64;
    float* acc_t_b = ws + off; off += (size_t)N_T * 64;
    float* deg_t_a = ws + off; off += (size_t)N_T;
    float* deg_t_b = ws + off; off += (size_t)N_T;
    float* acc_c   = ws + off; off += (size_t)N_C * 64;
    float* acc_d   = ws + off; off += (size_t)N_D * 64;
    float* deg_c   = ws + off; off += (size_t)N_C;
    float* deg_d   = ws + off; off += (size_t)N_D;
    float* Wh_t    = ws + off; off += (size_t)N_T * 64;  // features@W1[1] -> later h_t (L1)
    float* Wh_t2   = ws + off; off += (size_t)N_T * 64;  // features@W1[3]
    float* Wh_c    = ws + off; off += (size_t)N_C * 64;  // emb_card@W1[0] -> later h_c (L1)
    float* Wh_d    = ws + off; off += (size_t)N_D * 64;  // emb_device@W1[2] -> later h_d (L1)

    const long long zero1 = (long long)(acc_t_a ? 0 : 0) +
                            (long long)64 * (2LL * N_T + N_C + N_D) + (2LL * N_T + N_C + N_D);
    const long long zero2 = (long long)64 * 2LL * N_T + 2LL * N_T;

    dim3 blk(256);
    const int gT = cdiv_i(N_T, 128), gC = cdiv_i(N_C, 128), gD = cdiv_i(N_D, 128);
    const int gE = cdiv_i((long long)E * 16, 256);

    // ---------------- Layer 1 ----------------
    gemm64_wmma_kernel<<<gC, blk, 0, stream>>>(emb_card,   W1 + 0 * 4096, b1 + 0 * 64, Wh_c,  N_C);
    gemm64_wmma_kernel<<<gT, blk, 0, stream>>>(features,   W1 + 1 * 4096, b1 + 1 * 64, Wh_t,  N_T);
    gemm64_wmma_kernel<<<gD, blk, 0, stream>>>(emb_device, W1 + 2 * 4096, b1 + 2 * 64, Wh_d,  N_D);
    gemm64_wmma_kernel<<<gT, blk, 0, stream>>>(features,   W1 + 3 * 4096, b1 + 3 * 64, Wh_t2, N_T);

    zero_kernel<<<cdiv_i(zero1, 256), blk, 0, stream>>>(ws, zero1);

    scatter_acc_kernel<<<gE, blk, 0, stream>>>(Wh_c,  src_c2t, dst_c2t, acc_t_a, deg_t_a, E);
    scatter_acc_kernel<<<gE, blk, 0, stream>>>(Wh_d,  src_d2t, dst_d2t, acc_t_b, deg_t_b, E);
    scatter_acc_kernel<<<gE, blk, 0, stream>>>(Wh_t,  src_t2c, dst_t2c, acc_c,   deg_c,   E);
    scatter_acc_kernel<<<gE, blk, 0, stream>>>(Wh_t2, src_t2d, dst_t2d, acc_d,   deg_d,   E);

    // h_t/h_c/h_d overwrite the (now dead) Wh buffers.
    finalize_sum2_lrelu_kernel<<<cdiv_i((long long)N_T * 64, 256), blk, 0, stream>>>(
        acc_t_a, acc_t_b, deg_t_a, deg_t_b, Wh_t, (long long)N_T * 64);
    finalize_lrelu_kernel<<<cdiv_i((long long)N_C * 64, 256), blk, 0, stream>>>(
        acc_c, deg_c, Wh_c, (long long)N_C * 64);
    finalize_lrelu_kernel<<<cdiv_i((long long)N_D * 64, 256), blk, 0, stream>>>(
        acc_d, deg_d, Wh_d, (long long)N_D * 64);

    // ---------------- Layer 2 (target path only; h_c/h_d of L2 are unused) ----------------
    // GEMM outputs reuse the freed acc_c / acc_d regions.
    gemm64_wmma_kernel<<<gC, blk, 0, stream>>>(Wh_c, W2 + 0 * 4096, b2 + 0 * 64, acc_c, N_C);
    gemm64_wmma_kernel<<<gD, blk, 0, stream>>>(Wh_d, W2 + 2 * 4096, b2 + 2 * 64, acc_d, N_D);

    zero_kernel<<<cdiv_i(zero2, 256), blk, 0, stream>>>(ws, zero2);

    scatter_acc_kernel<<<gE, blk, 0, stream>>>(acc_c, src_c2t, dst_c2t, acc_t_a, deg_t_a, E);
    scatter_acc_kernel<<<gE, blk, 0, stream>>>(acc_d, src_d2t, dst_d2t, acc_t_b, deg_t_b, E);

    finalize_out_kernel<<<cdiv_i(N_T, 256), blk, 0, stream>>>(
        acc_t_a, acc_t_b, deg_t_a, deg_t_b, W_out, b_out, out, N_T);

    (void)n_in; (void)out_size; (void)ws_size;
}
